// LorentzTransformerBlock_53231824666658
// MI455X (gfx1250) — matrix operand order repaired and checked by reading, not compiled, not run on hardware
//
#include <hip/hip_runtime.h>
#include <hip/hip_bf16.h>
#include <stdint.h>

// ---------------------------------------------------------------------------
// CDNA5 (gfx1250) wave32 WMMA implementation of the Lorentz transformer block.
// ---------------------------------------------------------------------------

typedef __attribute__((ext_vector_type(16))) _Float16 v16h;
typedef __attribute__((ext_vector_type(8)))  float    v8f;
typedef __attribute__((ext_vector_type(4)))  _Float16 h4v;

#define K_CURV 1.0f
#define EPS_C  1e-6f

union HalfVec {
    v16h     v;
    uint32_t u[8];
};

// A fragment (16x32 f16, M x K), ISA 7.12.2 layout. Pair loads are contiguous
// 16B runs -> compiler merges to ds_load_b128 x2.
__device__ inline v16h a_frag_ld(const _Float16* rowp, int lane) {
    HalfVec f;
    const int kb = (lane & 16) ? 8 : 0;
#pragma unroll
    for (int r = 0; r < 8; ++r) {
        const int k0 = (r < 4) ? (kb + 2 * r) : (16 + kb + 2 * (r - 4));
        f.u[r] = *(const uint32_t*)(rowp + k0);
    }
    return f.v;
}

// B fragment (32x16 f16, K x N): lane holds column n = lane%16, K values
// kb..kb+7 and 16+kb..16+kb+7 (kb = 0 / 8), matching A. colp = column-contiguous LDS row.
__device__ inline v16h b_frag_ld(const _Float16* colp, int lane) {
    HalfVec f;
    const _Float16* p = colp + ((lane & 16) ? 8 : 0);
#pragma unroll
    for (int r = 0; r < 8; ++r) f.u[r] = *(const uint32_t*)(p + ((r < 4) ? 2 * r : 16 + 2 * (r - 4)));
    return f.v;
}

// ---------------------------------------------------------------------------
// Weight preprocessing: f32 W[Ka x N] -> f16, transposed + zero padded:
// out[n*Kpad + k] = W[k*N + n] (0 outside). Runs once per launch, tiny.
// ---------------------------------------------------------------------------
__global__ __launch_bounds__(256) void prep_weight_kernel(
    const float* __restrict__ W, _Float16* __restrict__ out, int Ka, int N,
    int Kpad) {
    const int idx = blockIdx.x * 256 + threadIdx.x;
    const int kk  = idx % Kpad;
    const int n   = idx / Kpad;
    const float v = (kk < Ka && n < N) ? W[(size_t)kk * N + n] : 0.f;
    out[idx] = (_Float16)v;
}

// ---------------------------------------------------------------------------
// Lorentz layernorm + projection (wave per row, 511 space elements).
// ---------------------------------------------------------------------------
__global__ __launch_bounds__(256) void ln_project_kernel(
    const float* __restrict__ x, const float* __restrict__ gamma,
    const float* __restrict__ beta, float* __restrict__ out, int Dm1, int Dfull) {
    const int lane = threadIdx.x & 31;
    const int wave = threadIdx.x >> 5;
    const int row  = blockIdx.x * 8 + wave;
    const float* xr = x + (size_t)row * Dfull + 1;

    float vals[16];
    float s = 0.f, s2 = 0.f;
#pragma unroll
    for (int i = 0; i < 16; ++i) {
        const int idx = lane + 32 * i;
        float vv = 0.f;
        if (idx < Dm1) vv = xr[idx];
        vals[i] = vv;
        s += vv; s2 += vv * vv;
    }
    for (int m = 1; m < 32; m <<= 1) { s += __shfl_xor(s, m); s2 += __shfl_xor(s2, m); }
    const float invn = 1.0f / (float)Dm1;
    const float mu   = s * invn;
    const float var  = s2 * invn - mu * mu;
    const float rstd = rsqrtf(var + 1e-5f);

    float nsq = 0.f;
    float* orow = out + (size_t)row * Dfull;
#pragma unroll
    for (int i = 0; i < 16; ++i) {
        const int idx = lane + 32 * i;
        if (idx < Dm1) {
            const float nv = (vals[i] - mu) * rstd * gamma[idx] + beta[idx];
            orow[1 + idx] = nv;
            nsq += nv * nv;
        }
    }
    for (int m = 1; m < 32; m <<= 1) nsq += __shfl_xor(nsq, m);
    if (lane == 0) orow[0] = sqrtf(fmaxf(K_CURV + nsq, EPS_C));
}

// ---------------------------------------------------------------------------
// Generic projection: buf[row][0] = sqrt(clip(K + sum(buf[row][1:]^2), eps))
// ---------------------------------------------------------------------------
__global__ __launch_bounds__(256) void project_kernel(float* __restrict__ buf,
                                                      int Dfull) {
    const int lane = threadIdx.x & 31;
    const int wave = threadIdx.x >> 5;
    const int row  = blockIdx.x * 8 + wave;
    float* r = buf + (size_t)row * Dfull;
    const int Dm1 = Dfull - 1;
    float ss = 0.f;
    for (int idx = lane; idx < Dm1; idx += 32) {
        const float vv = r[1 + idx];
        ss += vv * vv;
    }
    for (int m = 1; m < 32; m <<= 1) ss += __shfl_xor(ss, m);
    if (lane == 0) r[0] = sqrtf(fmaxf(K_CURV + ss, EPS_C));
}

// ---------------------------------------------------------------------------
// WMMA GEMM, double-buffered LDS, software-pipelined: global loads for tile
// kt+1 are issued BEFORE the WMMAs of tile kt and only consumed (convert +
// ds_store) after them, so loads overlap matrix math.
// A: f32 activations (M x Kpad exact). WT: f16 transposed padded weights
// (Npad x Kpad). out[m][n+1] = epilogue(A@W + bias).
//   mode 0: bias   mode 1: bias + tanh-gelu   mode 2: res + scale*(acc+bias)
// Grid: (Npad/128, M/128), 256 threads = 8 waves, each 32x64 (2x4 WMMA).
// ---------------------------------------------------------------------------
#define BM 128
#define BN 128
#define BK 32
#define LSTR 40  // padded LDS row stride in halves

__global__ __launch_bounds__(256) void gemm_wmma_kernel(
    const float* __restrict__ A, int lda, const _Float16* __restrict__ WT,
    int Kpad, const float* __restrict__ bias, float* __restrict__ out, int ldo,
    const float* __restrict__ res, int ldr, float scale, int N, int mode) {
    __shared__ _Float16 As[2][BM][LSTR];
    __shared__ _Float16 BsT[2][BN][LSTR];

    const int tid  = threadIdx.x;
    const int lane = tid & 31;
    const int wave = tid >> 5;
    const int bm   = blockIdx.y * BM;
    const int bn   = blockIdx.x * BN;
    const int wr   = (wave >> 1) * 32;
    const int wc   = (wave & 1) * 64;

    // Staging assignments (fixed per thread).
    const int ar0 = tid >> 3;          // A: rows ar0 + {0,32,64,96}
    const int ac4 = (tid & 7) << 2;    // A: 4-col group
    const int bnr = tid >> 1;          // B: one n row (0..127)
    const int bh  = (tid & 1) << 4;    // B: 16-half group (0 or 16)

    float4 ra[4];  // in-flight A tile
    uint4  rbv;    // in-flight B tile (f16 bits pass-through)

    auto loadTiles = [&](int k0g) {
#pragma unroll
        for (int s = 0; s < 4; ++s)
            ra[s] = *(const float4*)(A + (size_t)(bm + ar0 + 32 * s) * lda +
                                     (k0g + ac4));
        rbv = *(const uint4*)(WT + (size_t)(bn + bnr) * Kpad + (k0g + bh));
    };
    auto storeTiles = [&](int buf) {
#pragma unroll
        for (int s = 0; s < 4; ++s) {
            h4v hv;
            hv[0] = (_Float16)ra[s].x; hv[1] = (_Float16)ra[s].y;
            hv[2] = (_Float16)ra[s].z; hv[3] = (_Float16)ra[s].w;
            *(h4v*)&As[buf][ar0 + 32 * s][ac4] = hv;
        }
        *(uint4*)&BsT[buf][bnr][bh] = rbv;
    };

    const v8f vzero = {0.f, 0.f, 0.f, 0.f, 0.f, 0.f, 0.f, 0.f};
    v8f acc[2][4];
#pragma unroll
    for (int i = 0; i < 2; ++i)
#pragma unroll
        for (int j = 0; j < 4; ++j) acc[i][j] = vzero;

    const int ksteps = Kpad / BK;
    loadTiles(0);
    for (int kt = 0; kt < ksteps; ++kt) {
        const int cb = kt & 1;
        storeTiles(cb);  // consumes loads issued last iteration (post-WMMA)
        __syncthreads();
        if (kt + 1 < ksteps) loadTiles((kt + 1) * BK);  // issue only

        v16h af[2], bf[4];
#pragma unroll
        for (int i = 0; i < 2; ++i)
            af[i] = a_frag_ld(&As[cb][wr + i * 16 + (lane & 15)][0], lane);
#pragma unroll
        for (int j = 0; j < 4; ++j)
            bf[j] = b_frag_ld(&BsT[cb][wc + j * 16 + (lane & 15)][0], lane);
#pragma unroll
        for (int i = 0; i < 2; ++i)
#pragma unroll
            for (int j = 0; j < 4; ++j)
                acc[i][j] = __builtin_amdgcn_wmma_f32_16x16x32_f16(
                    false, af[i], false, bf[j], (short)0, acc[i][j], false, false);
    }

    // Epilogue: lane -> col n = lane%16; VGPR e -> row e (+8 if lane>=16).
    const int nloc  = lane & 15;
    const int rbase = (lane & 16) ? 8 : 0;
#pragma unroll
    for (int i = 0; i < 2; ++i) {
#pragma unroll
        for (int j = 0; j < 4; ++j) {
            const int gn = bn + wc + j * 16 + nloc;
            if (gn >= N) continue;
            const float bval = bias[gn];
#pragma unroll
            for (int e = 0; e < 8; ++e) {
                const int gm = bm + wr + i * 16 + rbase + e;
                float s = acc[i][j][e] + bval;
                if (mode == 1) {
                    const float x3 = s * s * s;
                    s = 0.5f * s *
                        (1.0f + tanhf(0.7978845608028654f * (s + 0.044715f * x3)));
                } else if (mode == 2) {
                    s = res[(size_t)gm * ldr + gn + 1] + scale * s;
                }
                out[(size_t)gm * ldo + gn + 1] = s;
            }
        }
    }
}

// ---------------------------------------------------------------------------
// Lorentz flash attention. Grid (S/128, H, B), 256 threads = 8 waves.
// attn = softmax(q~ . k~), q~ = 0.25*[-t_q, q_s], k~ = [t_k, k_s]; agg is
// Lorentz-normalized; 63 space dims/head written to outsp (M x 512 padded,
// head 7 lanes also zero the 8 pad columns). K/V tiles are preloaded into
// registers one step ahead so global loads overlap WMMA + softmax work.
// ---------------------------------------------------------------------------
__global__ __launch_bounds__(256) void lorentz_attn_kernel(
    const float* __restrict__ q, const float* __restrict__ k,
    const float* __restrict__ v, float* __restrict__ outsp, int S, int Dfull) {
    __shared__ _Float16 Qs[128][72];
    __shared__ _Float16 Ks[32][72];
    __shared__ _Float16 VT[64][40];
    __shared__ _Float16 Ps[8][16][40];

    const int tid   = threadIdx.x;
    const int lane  = tid & 31;
    const int wave  = tid >> 5;
    const int qb    = blockIdx.x;
    const int head  = blockIdx.y;
    const int batch = blockIdx.z;
    const size_t rowbase = (size_t)batch * S;
    const int hoff = head * 64;

    // ---- Stage Q once: 2 threads/row, unconditional unrolled loads ----
    {
        const int r    = tid >> 1;
        const int half = tid & 1;
        const int srow = qb * 128 + r;
        const float* qrow = q + (rowbase + srow) * Dfull + hoff;
        float ss = 0.f;
#pragma unroll
        for (int i = 0; i < 31; ++i) {
            const int d = 1 + half + 2 * i;  // <= 63
            const float val = qrow[d];
            ss += val * val;
            Qs[r][d] = (_Float16)(0.25f * val);
        }
        if (half == 0) {
            const float val = qrow[63];
            ss += val * val;
            Qs[r][63] = (_Float16)(0.25f * val);
        }
        ss += __shfl_xor(ss, 1);
        if (half == 0)
            Qs[r][0] = (_Float16)(-0.25f * sqrtf(fmaxf(K_CURV + ss, EPS_C)));
    }
    __syncthreads();

    const int wq = wave * 16;
    v16h aq[2];
    aq[0] = a_frag_ld(&Qs[wq + (lane & 15)][0], lane);
    aq[1] = a_frag_ld(&Qs[wq + (lane & 15)][32], lane);

    // K/V staging assignment: 8 threads per key.
    const int key = tid >> 3;
    const int j8  = tid & 7;
    float kreg[8], vreg[8];  // in-flight K/V tile (this thread's 8 dims)

    auto preloadKV = [&](int kt) {
        const int srow = kt * 32 + key;
        const float* krow = k + (rowbase + srow) * Dfull + hoff;
        const float* vrow = v + (rowbase + srow) * Dfull + hoff;
#pragma unroll
        for (int i = 0; i < 7; ++i) {
            const int d = 1 + j8 + 8 * i;  // <= 56
            kreg[i] = krow[d];
            vreg[i] = vrow[d];
        }
        if (j8 < 7) {
            const int d = 57 + j8;  // 57..63
            kreg[7] = krow[d];
            vreg[7] = vrow[d];
        } else {
            kreg[7] = 0.f;
            vreg[7] = 0.f;
        }
    };
    auto storeKV = [&]() {
        float ssk = 0.f, ssv = 0.f;
#pragma unroll
        for (int i = 0; i < 7; ++i) {
            const int d = 1 + j8 + 8 * i;
            ssk += kreg[i] * kreg[i];
            ssv += vreg[i] * vreg[i];
            Ks[key][d] = (_Float16)kreg[i];
            VT[d][key] = (_Float16)vreg[i];
        }
        ssk += kreg[7] * kreg[7];  // zero when j8 == 7
        ssv += vreg[7] * vreg[7];
        if (j8 < 7) {
            const int d = 57 + j8;
            Ks[key][d] = (_Float16)kreg[7];
            VT[d][key] = (_Float16)vreg[7];
        }
        ssk += __shfl_xor(ssk, 1); ssk += __shfl_xor(ssk, 2); ssk += __shfl_xor(ssk, 4);
        ssv += __shfl_xor(ssv, 1); ssv += __shfl_xor(ssv, 2); ssv += __shfl_xor(ssv, 4);
        if (j8 == 0) {
            Ks[key][0] = (_Float16)sqrtf(fmaxf(K_CURV + ssk, EPS_C));
            VT[0][key] = (_Float16)sqrtf(fmaxf(K_CURV + ssv, EPS_C));
        }
    };

    const v8f vzero = {0.f, 0.f, 0.f, 0.f, 0.f, 0.f, 0.f, 0.f};
    v8f o[4];
#pragma unroll
    for (int j = 0; j < 4; ++j) o[j] = vzero;
    float rmax[8], rsum[8];
#pragma unroll
    for (int e = 0; e < 8; ++e) { rmax[e] = -1e30f; rsum[e] = 0.f; }

    const int nsteps = S / 32;
    preloadKV(0);
    for (int kt = 0; kt < nsteps; ++kt) {
        __syncthreads();  // all waves done reading Ks/VT of previous step
        storeKV();
        __syncthreads();
        if (kt + 1 < nsteps) preloadKV(kt + 1);  // issue only; waits next iter

        // ---- Scores: two 16x16 tiles over d=64 ----
        v8f s0 = vzero, s1 = vzero;
        {
            const v16h b00 = b_frag_ld(&Ks[(lane & 15)][0], lane);
            const v16h b01 = b_frag_ld(&Ks[(lane & 15)][32], lane);
            const v16h b10 = b_frag_ld(&Ks[16 + (lane & 15)][0], lane);
            const v16h b11 = b_frag_ld(&Ks[16 + (lane & 15)][32], lane);
            s0 = __builtin_amdgcn_wmma_f32_16x16x32_f16(false, aq[0], false, b00, (short)0, s0, false, false);
            s0 = __builtin_amdgcn_wmma_f32_16x16x32_f16(false, aq[1], false, b01, (short)0, s0, false, false);
            s1 = __builtin_amdgcn_wmma_f32_16x16x32_f16(false, aq[0], false, b10, (short)0, s1, false, false);
            s1 = __builtin_amdgcn_wmma_f32_16x16x32_f16(false, aq[1], false, b11, (short)0, s1, false, false);
        }

        // ---- Streaming softmax ----
        float p0[8], p1[8];
#pragma unroll
        for (int e = 0; e < 8; ++e) {
            float cm = fmaxf(s0[e], s1[e]);
            cm = fmaxf(cm, __shfl_xor(cm, 1));
            cm = fmaxf(cm, __shfl_xor(cm, 2));
            cm = fmaxf(cm, __shfl_xor(cm, 4));
            cm = fmaxf(cm, __shfl_xor(cm, 8));
            const float mnew = fmaxf(rmax[e], cm);
            const float corr = __expf(rmax[e] - mnew);
            rmax[e] = mnew;
            p0[e] = __expf(s0[e] - mnew);
            p1[e] = __expf(s1[e] - mnew);
            float ps = p0[e] + p1[e];
            ps += __shfl_xor(ps, 1); ps += __shfl_xor(ps, 2);
            ps += __shfl_xor(ps, 4); ps += __shfl_xor(ps, 8);
            rsum[e] = rsum[e] * corr + ps;
#pragma unroll
            for (int j = 0; j < 4; ++j) o[j][e] *= corr;
        }

        // ---- Route P through per-wave LDS to build A fragment ----
        {
            const int rb = (lane & 16) ? 8 : 0;
            const int c  = lane & 15;
#pragma unroll
            for (int e = 0; e < 8; ++e) {
                Ps[wave][rb + e][c]      = (_Float16)p0[e];
                Ps[wave][rb + e][16 + c] = (_Float16)p1[e];
            }
        }
        const v16h ap = a_frag_ld(&Ps[wave][(lane & 15)][0], lane);
#pragma unroll
        for (int j = 0; j < 4; ++j) {
            const v16h bv = b_frag_ld(&VT[j * 16 + (lane & 15)][0], lane);
            o[j] = __builtin_amdgcn_wmma_f32_16x16x32_f16(false, ap, false, bv, (short)0, o[j], false, false);
        }
    }

    // ---- Epilogue: softmax + Lorentz normalize, write space, pad cols ----
    const int nloc = lane & 15;
    const int rb   = (lane & 16) ? 8 : 0;
#pragma unroll
    for (int e = 0; e < 8; ++e) {
        const float inv = 1.0f / rsum[e];
        float agg[4];
        float ssq = 0.f, t0 = 0.f;
#pragma unroll
        for (int j = 0; j < 4; ++j) {
            agg[j] = o[j][e] * inv;
            const int d = j * 16 + nloc;
            if (d == 0) t0 = agg[j];
            else        ssq += agg[j] * agg[j];
        }
        ssq += __shfl_xor(ssq, 1); ssq += __shfl_xor(ssq, 2);
        ssq += __shfl_xor(ssq, 4); ssq += __shfl_xor(ssq, 8);
        t0 += __shfl_xor(t0, 1); t0 += __shfl_xor(t0, 2);
        t0 += __shfl_xor(t0, 4); t0 += __shfl_xor(t0, 8);
        const float den  = sqrtf(fmaxf(fabsf(ssq - t0 * t0), EPS_C));
        const float invd = 1.0f / den;  // sqrt(K)=1
        const int srow = qb * 128 + wq + rb + e;
        float* orow_full = outsp + (rowbase + srow) * 512;
        float* orow      = orow_full + head * 63;
#pragma unroll
        for (int j = 0; j < 4; ++j) {
            const int d = j * 16 + nloc;
            if (d >= 1) orow[d - 1] = agg[j] * invd;
        }
        if (head == 7 && nloc < 8) orow_full[504 + nloc] = 0.f;  // pad cols
    }
}

// ---------------------------------------------------------------------------
// Host-side orchestration.
// ---------------------------------------------------------------------------
extern "C" void kernel_launch(void* const* d_in, const int* in_sizes, int n_in,
                              void* d_out, int out_size, void* d_ws, size_t ws_size,
                              hipStream_t stream) {
    const float* x   = (const float*)d_in[0];
    const float* Wq  = (const float*)d_in[1];
    const float* bq  = (const float*)d_in[2];
    const float* Wk  = (const float*)d_in[3];
    const float* bk  = (const float*)d_in[4];
    const float* Wv  = (const float*)d_in[5];
    const float* bv  = (const float*)d_in[6];
    const float* Wo  = (const float*)d_in[7];
    const float* bo  = (const float*)d_in[8];
    const float* g1  = (const float*)d_in[9];
    const float* be1 = (const float*)d_in[10];
    const float* g2  = (const float*)d_in[11];
    const float* be2 = (const float*)d_in[12];
    const float* W1  = (const float*)d_in[13];
    const float* b1  = (const float*)d_in[14];
    const float* W2  = (const float*)d_in[15];
    const float* b2  = (const float*)d_in[16];
    (void)in_sizes; (void)n_in; (void)out_size; (void)ws_size;

    const int B = 4, S = 1024, D = 512, HID = 2048;
    const int M = B * S;

    float* ws   = (float*)d_ws;
    float* h    = ws;
    float* qb_  = h   + (size_t)M * D;
    float* kb_  = qb_ + (size_t)M * D;
    float* vb_  = kb_ + (size_t)M * D;
    float* x1   = vb_ + (size_t)M * D;
    float* h2   = x1  + (size_t)M * D;
    float* asp  = h2  + (size_t)M * D;       // M x 512 (504 + zeroed pad)
    float* mbuf = asp + (size_t)M * D;       // M x 2048
    _Float16* wq16 = (_Float16*)(mbuf + (size_t)M * HID);
    _Float16* wk16 = wq16 + 512 * 512;
    _Float16* wv16 = wk16 + 512 * 512;
    _Float16* wo16 = wv16 + 512 * 512;
    _Float16* w116 = wo16 + 512 * 512;   // Npad 2048 x Kpad 512
    _Float16* w216 = w116 + 2048 * 512;  // Npad 512  x Kpad 2048
    float* outp = (float*)d_out;

    dim3 blk(256);

    // 0) weights -> f16, transposed (Npad x Kpad), zero-padded
    prep_weight_kernel<<<(512 * 512) / 256, blk, 0, stream>>>(Wq, wq16, 512, 511, 512);
    prep_weight_kernel<<<(512 * 512) / 256, blk, 0, stream>>>(Wk, wk16, 512, 511, 512);
    prep_weight_kernel<<<(512 * 512) / 256, blk, 0, stream>>>(Wv, wv16, 512, 511, 512);
    prep_weight_kernel<<<(512 * 512) / 256, blk, 0, stream>>>(Wo, wo16, 504, 511, 512);
    prep_weight_kernel<<<(2048 * 512) / 256, blk, 0, stream>>>(W1, w116, 512, 2047, 512);
    prep_weight_kernel<<<(2048 * 512) / 256, blk, 0, stream>>>(W2, w216, 2048, 511, 2048);

    // 1) h = lorentz_layernorm(x)
    ln_project_kernel<<<M / 8, blk, 0, stream>>>(x, g1, be1, h, D - 1, D);
    // 2) q/k/v = h @ W + b (space at cols 1..511)
    gemm_wmma_kernel<<<dim3(4, 32), blk, 0, stream>>>(h, D, wq16, 512, bq, qb_, D, nullptr, 0, 0.f, D - 1, 0);
    gemm_wmma_kernel<<<dim3(4, 32), blk, 0, stream>>>(h, D, wk16, 512, bk, kb_, D, nullptr, 0, 0.f, D - 1, 0);
    gemm_wmma_kernel<<<dim3(4, 32), blk, 0, stream>>>(h, D, wv16, 512, bv, vb_, D, nullptr, 0, 0.f, D - 1, 0);
    // 3) attention -> asp (M x 512, pad cols zeroed)
    lorentz_attn_kernel<<<dim3(S / 128, 8, B), blk, 0, stream>>>(qb_, kb_, vb_, asp, S, D);
    // 4) x1_space = x_space + 27.5*(asp @ Wo + bo)
    gemm_wmma_kernel<<<dim3(4, 32), blk, 0, stream>>>(asp, D, wo16, 512, bo, x1, D, x, D, 27.5f, D - 1, 2);
    // 5) h2 = lorentz_layernorm(x1)
    ln_project_kernel<<<M / 8, blk, 0, stream>>>(x1, g2, be2, h2, D - 1, D);
    // 6) mbuf_space = gelu(h2 @ W1 + b1); 7) mbuf time
    gemm_wmma_kernel<<<dim3(16, 32), blk, 0, stream>>>(h2, D, w116, 512, b1, mbuf, HID, nullptr, 0, 0.f, HID - 1, 1);
    project_kernel<<<M / 8, blk, 0, stream>>>(mbuf, HID);
    // 8) out_space = x1_space + 27.5*(mbuf @ W2 + b2); 9) out time
    gemm_wmma_kernel<<<dim3(4, 32), blk, 0, stream>>>(mbuf, HID, w216, 2048, b2, outp, D, x1, D, 27.5f, D - 1, 2);
    project_kernel<<<M / 8, blk, 0, stream>>>(outp, D);
}
